// RegionGatedMultiHeadAttention_13391708029710
// MI455X (gfx1250) — matrix operand order repaired and checked by reading, not compiled, not run on hardware
//
#include <hip/hip_runtime.h>

// ---------------------------------------------------------------------------
// RegionGatedMultiHeadAttention for MI455X (gfx1250, wave32, WMMA)
// B=2, N=2048, D=768, H=12, HD=64.
// Register-blocked: GEMMs use 32x32 tiles/wave (4 WMMA acc, 2x fragment
// reuse); attention processes 32 queries/wave (2x reuse of K/V fragments).
// ---------------------------------------------------------------------------

constexpr int BATCH = 2;
constexpr int NTOK  = 2048;
constexpr int DM    = 768;
constexpr int NH    = 12;
constexpr int HD    = 64;
constexpr float SCL = 0.125f;  // 64^-0.5

typedef _Float16 v16h __attribute__((ext_vector_type(16)));
typedef _Float16 v8h  __attribute__((ext_vector_type(8)));
typedef float    v8f  __attribute__((ext_vector_type(8)));

union H16 { v16h v; v8h h[2]; };

__device__ inline v8f wmma_f16(v16h a, v16h b, v8f c) {
  return __builtin_amdgcn_wmma_f32_16x16x32_f16(false, a, false, b, (short)0, c,
                                                false, false);
}
__device__ inline v8f zero8() { return (v8f){0.f,0.f,0.f,0.f,0.f,0.f,0.f,0.f}; }

// ---------------------------------------------------------------------------
// f32 -> f16 elementwise convert
// ---------------------------------------------------------------------------
__global__ void convert_f16_kernel(const float* __restrict__ in,
                                   _Float16* __restrict__ out, int n) {
  int i = blockIdx.x * blockDim.x + threadIdx.x;
  if (i < n) out[i] = (_Float16)in[i];
}

// ---------------------------------------------------------------------------
// w[K][C] f32 -> wt[C][K] f16  (GEMM B-fragments then read contiguous K)
// ---------------------------------------------------------------------------
__global__ void transpose_f16_kernel(const float* __restrict__ w,
                                     _Float16* __restrict__ wt, int K, int C) {
  int i = blockIdx.x * blockDim.x + threadIdx.x;
  if (i >= K * C) return;
  int c = i % C;
  int k = i / C;
  wt[(size_t)c * K + k] = (_Float16)w[i];
}

// ---------------------------------------------------------------------------
// QKV epilogue helper: scatter one 16x16 f32 tile into Q/K (row-major) or
// V-transposed f16 buffers, adding bias.
// ---------------------------------------------------------------------------
__device__ inline void store_qkv_tile(const v8f& acc, int row0, int c,
                                      const float* __restrict__ bias,
                                      _Float16* __restrict__ Qh,
                                      _Float16* __restrict__ Kh,
                                      _Float16* __restrict__ Vt, int hi8) {
  const float bv = bias[c];
  const int s   = c / DM;          // 0=q 1=k 2=v
  const int rem = c % DM;
  const int h   = rem >> 6;
  const int d   = rem & 63;
  if (s == 2) {
    const int rbase = row0 + hi8;
    const int bb = rbase >> 11;
    const int n0 = rbase & (NTOK - 1);
    v8h pack;
#pragma unroll
    for (int r = 0; r < 8; ++r) pack[r] = (_Float16)(acc[r] + bv);
    *(v8h*)(Vt + ((size_t)(bb * NH + h) * HD + d) * NTOK + n0) = pack;
  } else {
    _Float16* base = (s == 0) ? Qh : Kh;
#pragma unroll
    for (int r = 0; r < 8; ++r) {
      const int row = row0 + hi8 + r;
      const int bb = row >> 11;
      const int n = row & (NTOK - 1);
      base[((size_t)(bb * NH + h) * NTOK + n) * HD + d] = (_Float16)(acc[r] + bv);
    }
  }
}

// ---------------------------------------------------------------------------
// QKV GEMM: C[4096 x 2304] = xh * wT^T + bias, 32x32 tile per wave.
// ---------------------------------------------------------------------------
__global__ void __launch_bounds__(128) qkv_gemm_kernel(
    const _Float16* __restrict__ xh, const _Float16* __restrict__ wT,
    const float* __restrict__ bias, _Float16* __restrict__ Qh,
    _Float16* __restrict__ Kh, _Float16* __restrict__ Vt) {
  const int wid  = (blockIdx.x * blockDim.x + threadIdx.x) >> 5;
  const int lane = threadIdx.x & 31;
  const int TN = (3 * DM) / 32;             // 72 col tiles
  const int tm = wid / TN, tn = wid % TN;
  if (tm >= (BATCH * NTOK) / 32) return;

  const int l15  = lane & 15;
  const int hi8  = (lane & 16) ? 8 : 0;
  const int hi16 = (lane & 16) ? 16 : 0;

  const _Float16* ap0 = xh + (size_t)(tm * 32 + l15) * DM;
  const _Float16* ap1 = ap0 + (size_t)16 * DM;
  const _Float16* bp0 = wT + (size_t)(tn * 32 + l15) * DM;
  const _Float16* bp1 = bp0 + (size_t)16 * DM;

  v8f a00 = zero8(), a01 = zero8(), a10 = zero8(), a11 = zero8();
  for (int k0 = 0; k0 < DM; k0 += 32) {
    H16 a0, a1, b0, b1;
    a0.h[0] = *(const v8h*)(ap0 + k0 + hi8);
    a0.h[1] = *(const v8h*)(ap0 + k0 + 16 + hi8);
    a1.h[0] = *(const v8h*)(ap1 + k0 + hi8);
    a1.h[1] = *(const v8h*)(ap1 + k0 + 16 + hi8);
    b0.h[0] = *(const v8h*)(bp0 + k0 + hi8);
    b0.h[1] = *(const v8h*)(bp0 + k0 + 16 + hi8);
    b1.h[0] = *(const v8h*)(bp1 + k0 + hi8);
    b1.h[1] = *(const v8h*)(bp1 + k0 + 16 + hi8);
    a00 = wmma_f16(a0.v, b0.v, a00);
    a01 = wmma_f16(a0.v, b1.v, a01);
    a10 = wmma_f16(a1.v, b0.v, a10);
    a11 = wmma_f16(a1.v, b1.v, a11);
  }

  const int r0 = tm * 32, c0 = tn * 32 + l15;
  store_qkv_tile(a00, r0,      c0,      bias, Qh, Kh, Vt, hi8);
  store_qkv_tile(a01, r0,      c0 + 16, bias, Qh, Kh, Vt, hi8);
  store_qkv_tile(a10, r0 + 16, c0,      bias, Qh, Kh, Vt, hi8);
  store_qkv_tile(a11, r0 + 16, c0 + 16, bias, Qh, Kh, Vt, hi8);
}

// ---------------------------------------------------------------------------
// Region-gated flash attention. One wave per (b, h, 32-query tile).
// S^T = K_tile x Q^T: lane == query column -> softmax is per-lane + one
// shfl_xor(16). Two query groups (u=0,1) share all K/V fragments and mask
// loads: 16 WMMAs per 32-key chunk at ~32 FLOP/byte.
// ---------------------------------------------------------------------------
__global__ void __launch_bounds__(128) attn_kernel(
    const _Float16* __restrict__ Qh, const _Float16* __restrict__ Kh,
    const _Float16* __restrict__ Vt, const float* __restrict__ masks,
    const float* __restrict__ lambda_r, const float* __restrict__ theta_r,
    const float* __restrict__ mask_w, _Float16* __restrict__ attn_out) {
  const int wid  = (blockIdx.x * blockDim.x + threadIdx.x) >> 5;
  const int lane = threadIdx.x & 31;
  const int qt = wid & 63;        // 64 query tiles of 32
  const int bh = wid >> 6;        // 0..23
  if (bh >= BATCH * NH) return;
  const int b = bh / NH;
  const int h = bh % NH;

  const int l15  = lane & 15;
  const int hi8  = (lane & 16) ? 8 : 0;
  const int hi16 = (lane & 16) ? 16 : 0;
  const int qb = qt * 32;

  const float lam   = lambda_r[bh];
  const float theta = theta_r[bh];
  const float mw    = mask_w[bh];
  const float* mkbase = masks + (size_t)bh * NTOK;
  const float mq[2] = {mkbase[qb + l15], mkbase[qb + 16 + l15]};

  // Preload Q B-fragments for both query groups (channels 0-31, 32-63).
  H16 qf[2][2];
#pragma unroll
  for (int u = 0; u < 2; ++u) {
    const _Float16* qptr = Qh + ((size_t)bh * NTOK + qb + u * 16 + l15) * HD;
    qf[u][0].h[0] = *(const v8h*)(qptr + hi8);
    qf[u][0].h[1] = *(const v8h*)(qptr + 16 + hi8);
    qf[u][1].h[0] = *(const v8h*)(qptr + 32 + hi8);
    qf[u][1].h[1] = *(const v8h*)(qptr + 48 + hi8);
  }

  const _Float16* kbase = Kh + (size_t)bh * NTOK * HD;
  const _Float16* vbase = Vt + (size_t)bh * HD * NTOK;

  float m[2] = {-1e30f, -1e30f};
  float l[2] = {0.f, 0.f};
  v8f O[2][4];
#pragma unroll
  for (int u = 0; u < 2; ++u)
#pragma unroll
    for (int g = 0; g < 4; ++g) O[u][g] = zero8();

  for (int kb = 0; kb < NTOK; kb += 32) {
    // ---- K A-fragments (shared by both query groups) ----
    H16 ka[2][2];
#pragma unroll
    for (int t = 0; t < 2; ++t) {
      const _Float16* kp = kbase + (size_t)(kb + t * 16 + l15) * HD;
      ka[t][0].h[0] = *(const v8h*)(kp + hi8);
      ka[t][0].h[1] = *(const v8h*)(kp + 16 + hi8);
      ka[t][1].h[0] = *(const v8h*)(kp + 32 + hi8);
      ka[t][1].h[1] = *(const v8h*)(kp + 48 + hi8);
    }
    // ---- key-mask values (shared) ----
    float mk[2][8];
#pragma unroll
    for (int t = 0; t < 2; ++t) {
      const float4 m0 = *(const float4*)(mkbase + kb + t * 16 + hi8);
      const float4 m1 = *(const float4*)(mkbase + kb + t * 16 + hi8 + 4);
      mk[t][0] = m0.x; mk[t][1] = m0.y; mk[t][2] = m0.z; mk[t][3] = m0.w;
      mk[t][4] = m1.x; mk[t][5] = m1.y; mk[t][6] = m1.z; mk[t][7] = m1.w;
    }

    H16 pfrag[2];
    float arow[2][8];
#pragma unroll
    for (int u = 0; u < 2; ++u) {
      // scores^T: two 16x16 tiles, accumulate over 64 channels
      v8f st[2];
      st[0] = wmma_f16(ka[0][0].v, qf[u][0].v, zero8());
      st[0] = wmma_f16(ka[0][1].v, qf[u][1].v, st[0]);
      st[1] = wmma_f16(ka[1][0].v, qf[u][0].v, zero8());
      st[1] = wmma_f16(ka[1][1].v, qf[u][1].v, st[1]);

      // gate + scale, chunk max over this lane's 16 keys
      float cmax = -3.0e38f;
#pragma unroll
      for (int t = 0; t < 2; ++t)
#pragma unroll
        for (int r = 0; r < 8; ++r) {
          const float g = 1.f / (1.f + __expf(-lam * (mq[u] * mk[t][r] - theta)));
          const float sv = st[t][r] * (SCL * g);
          st[t][r] = sv;
          cmax = fmaxf(cmax, sv);
        }
      // lanes L / L+16 split each query's keys: share max
      cmax = fmaxf(cmax, __shfl_xor(cmax, 16, 32));
      const float mnew  = fmaxf(m[u], cmax);
      const float alpha = __expf(m[u] - mnew);
      m[u] = mnew;

      float lsum = 0.f;
#pragma unroll
      for (int t = 0; t < 2; ++t)
#pragma unroll
        for (int r = 0; r < 8; ++r) {
          const float p = __expf(st[t][r] - mnew);
          lsum += p;
          pfrag[u].h[t][r] = (_Float16)p;  // packs directly as A-fragment
        }
      l[u] = l[u] * alpha + lsum;
#pragma unroll
      for (int r = 0; r < 8; ++r) arow[u][r] = __shfl(alpha, r + hi8, 32);
    }

    // ---- P x V: each V fragment reused by both query groups ----
#pragma unroll
    for (int g = 0; g < 4; ++g) {
      const _Float16* vp = vbase + (size_t)(g * 16 + l15) * NTOK + kb + hi8;
      H16 vf;
      vf.h[0] = *(const v8h*)(vp);
      vf.h[1] = *(const v8h*)(vp + 16);
#pragma unroll
      for (int u = 0; u < 2; ++u) {
#pragma unroll
        for (int r = 0; r < 8; ++r) O[u][g][r] *= arow[u][r];
        O[u][g] = wmma_f16(pfrag[u].v, vf.v, O[u][g]);
      }
    }
  }

  // ---- finalize: 1/l and per-head mask_weight, store f16 row-major ----
#pragma unroll
  for (int u = 0; u < 2; ++u) {
    const float ltot = l[u] + __shfl_xor(l[u], 16, 32);
    float rs[8];
#pragma unroll
    for (int r = 0; r < 8; ++r) rs[r] = mw / __shfl(ltot, r + hi8, 32);
#pragma unroll
    for (int g = 0; g < 4; ++g)
#pragma unroll
      for (int r = 0; r < 8; ++r) {
        const int row = qb + u * 16 + hi8 + r;        // token index
        const int dim = (g << 4) + l15;               // dim within head
        attn_out[((size_t)(b * NTOK + row)) * DM + h * HD + dim] =
            (_Float16)(O[u][g][r] * rs[r]);
      }
  }
}

// ---------------------------------------------------------------------------
// Output projection: out[4096 x 768] = Ah * wT^T + b, 32x32 tile per wave.
// ---------------------------------------------------------------------------
__global__ void __launch_bounds__(128) proj_gemm_kernel(
    const _Float16* __restrict__ ah, const _Float16* __restrict__ wT,
    const float* __restrict__ bias, float* __restrict__ out) {
  const int wid  = (blockIdx.x * blockDim.x + threadIdx.x) >> 5;
  const int lane = threadIdx.x & 31;
  const int TN = DM / 32;                       // 24
  const int tm = wid / TN, tn = wid % TN;
  if (tm >= (BATCH * NTOK) / 32) return;

  const int l15  = lane & 15;
  const int hi8  = (lane & 16) ? 8 : 0;
  const int hi16 = (lane & 16) ? 16 : 0;

  const _Float16* ap0 = ah + (size_t)(tm * 32 + l15) * DM;
  const _Float16* ap1 = ap0 + (size_t)16 * DM;
  const _Float16* bp0 = wT + (size_t)(tn * 32 + l15) * DM;
  const _Float16* bp1 = bp0 + (size_t)16 * DM;

  v8f a00 = zero8(), a01 = zero8(), a10 = zero8(), a11 = zero8();
  for (int k0 = 0; k0 < DM; k0 += 32) {
    H16 a0, a1, b0, b1;
    a0.h[0] = *(const v8h*)(ap0 + k0 + hi8);
    a0.h[1] = *(const v8h*)(ap0 + k0 + 16 + hi8);
    a1.h[0] = *(const v8h*)(ap1 + k0 + hi8);
    a1.h[1] = *(const v8h*)(ap1 + k0 + 16 + hi8);
    b0.h[0] = *(const v8h*)(bp0 + k0 + hi8);
    b0.h[1] = *(const v8h*)(bp0 + k0 + 16 + hi8);
    b1.h[0] = *(const v8h*)(bp1 + k0 + hi8);
    b1.h[1] = *(const v8h*)(bp1 + k0 + 16 + hi8);
    a00 = wmma_f16(a0.v, b0.v, a00);
    a01 = wmma_f16(a0.v, b1.v, a01);
    a10 = wmma_f16(a1.v, b0.v, a10);
    a11 = wmma_f16(a1.v, b1.v, a11);
  }

  const int c0 = tn * 32 + l15;
  const float bv0 = bias[c0], bv1 = bias[c0 + 16];
#pragma unroll
  for (int r = 0; r < 8; ++r) {
    const int row0 = tm * 32 + hi8 + r;
    out[(size_t)row0 * DM + c0]            = a00[r] + bv0;
    out[(size_t)row0 * DM + c0 + 16]       = a01[r] + bv1;
    out[(size_t)(row0 + 16) * DM + c0]      = a10[r] + bv0;
    out[(size_t)(row0 + 16) * DM + c0 + 16] = a11[r] + bv1;
  }
}

// ---------------------------------------------------------------------------
extern "C" void kernel_launch(void* const* d_in, const int* in_sizes, int n_in,
                              void* d_out, int out_size, void* d_ws,
                              size_t ws_size, hipStream_t stream) {
  (void)in_sizes; (void)n_in; (void)out_size; (void)ws_size;

  const float* x        = (const float*)d_in[0];
  const float* masks    = (const float*)d_in[1];
  const float* lambda_r = (const float*)d_in[2];
  const float* theta_r  = (const float*)d_in[3];
  const float* mask_w   = (const float*)d_in[4];
  const float* qkv_w    = (const float*)d_in[5];
  const float* qkv_b    = (const float*)d_in[6];
  const float* proj_w   = (const float*)d_in[7];
  const float* proj_b   = (const float*)d_in[8];
  float* out = (float*)d_out;

  // Workspace carve-up (f16 elements; total ~36.2 MB)
  _Float16* ws = (_Float16*)d_ws;
  size_t off = 0;
  _Float16* xh      = ws + off; off += (size_t)BATCH * NTOK * DM;
  _Float16* qkv_wT  = ws + off; off += (size_t)3 * DM * DM;
  _Float16* proj_wT = ws + off; off += (size_t)DM * DM;
  _Float16* Qh      = ws + off; off += (size_t)BATCH * NH * NTOK * HD;
  _Float16* Kh      = ws + off; off += (size_t)BATCH * NH * NTOK * HD;
  _Float16* Vt      = ws + off; off += (size_t)BATCH * NH * NTOK * HD;
  _Float16* Ah      = ws + off; off += (size_t)BATCH * NTOK * DM;

  // 1) precision/layout conversions
  {
    const int n = BATCH * NTOK * DM;
    convert_f16_kernel<<<(n + 255) / 256, 256, 0, stream>>>(x, xh, n);
  }
  {
    const int n = DM * 3 * DM;
    transpose_f16_kernel<<<(n + 255) / 256, 256, 0, stream>>>(qkv_w, qkv_wT, DM,
                                                              3 * DM);
  }
  {
    const int n = DM * DM;
    transpose_f16_kernel<<<(n + 255) / 256, 256, 0, stream>>>(proj_w, proj_wT,
                                                              DM, DM);
  }

  // 2) QKV projection: 128x72 = 9216 wave-tiles, 4 waves/block
  qkv_gemm_kernel<<<2304, 128, 0, stream>>>(xh, qkv_wT, qkv_b, Qh, Kh, Vt);

  // 3) region-gated flash attention: 24x64 = 1536 waves
  attn_kernel<<<384, 128, 0, stream>>>(Qh, Kh, Vt, masks, lambda_r, theta_r,
                                       mask_w, Ah);

  // 4) output projection: 128x24 = 3072 wave-tiles
  proj_gemm_kernel<<<768, 128, 0, stream>>>(Ah, proj_wT, proj_b, out);
}
